// CAT_81269371175150
// MI455X (gfx1250) — matrix operand order repaired and checked by reading, not compiled, not run on hardware
//
#include <hip/hip_runtime.h>
#include <hip/hip_bf16.h>
#include <math.h>

// Problem constants (match reference)
#define N_NODES 50000
#define F_IN    500
#define D_HID   256
#define K_CL    16
#define E_NNZ   800000

typedef __attribute__((ext_vector_type(16))) __bf16 v16bf;
typedef __attribute__((ext_vector_type(8)))  float  v8f;

union BF16Frag { v16bf v; unsigned short u[16]; unsigned int w[8]; uint4 q[2]; };

__device__ inline unsigned int pack2bf(float lo, float hi) {
  __hip_bfloat162 h = __float22bfloat162_rn(make_float2(lo, hi));
  return *reinterpret_cast<unsigned int*>(&h);
}

__device__ inline float warpSum(float v) {
  #pragma unroll
  for (int o = 16; o > 0; o >>= 1) v += __shfl_down(v, o);
  return v;
}

__device__ inline void atomicMaxF(float* addr, float val) {
  int* ai = (int*)addr;
  int old = __float_as_int(*addr);
  while (__int_as_float(old) < val) {
    int assumed = old;
    old = atomicCAS(ai, assumed, __float_as_int(val));
    if (old == assumed) break;
  }
}

// ---------------------------------------------------------------- fill
__global__ void fill_kernel(float* __restrict__ p, size_t n, float v) {
  size_t i = (size_t)blockIdx.x * blockDim.x + threadIdx.x;
  size_t stride = (size_t)gridDim.x * blockDim.x;
  for (; i < n; i += stride) p[i] = v;
}

// ---------------------------------------------------------------- GEMM: H = X @ W^T  (X:[n,500], W:[256,500], H:[n,256])
// 128x128 tile per block, 8 waves, each wave: 32x64 via 2x4 WMMA 16x16x32 bf16.
// Branch-free staging: row index clamped (garbage rows never stored), K tail peeled.
__global__ __launch_bounds__(256) void gemm_bf16_wmma(
    const float* __restrict__ X, const float* __restrict__ W,
    float* __restrict__ H, int nrows) {
  __shared__ unsigned short lA[128][40];  // [m][k] (k<32 used), 80B row = 16B aligned
  __shared__ unsigned short lB[128][40];  // [n][k]

  const int tid  = threadIdx.x;
  const int lane = tid & 31;
  const int w    = tid >> 5;
  const int wm   = w & 3;        // wave M index (0..3) -> 32 rows
  const int wn   = w >> 2;       // wave N index (0..1) -> 64 cols
  const int rowBase = blockIdx.x * 128;
  const int colBase = blockIdx.y * 128;

  v8f acc[2][4];
  #pragma unroll
  for (int i = 0; i < 2; i++)
    #pragma unroll
    for (int j = 0; j < 4; j++)
      acc[i][j] = (v8f){0.f,0.f,0.f,0.f,0.f,0.f,0.f,0.f};

  const int lr = tid >> 1;          // 0..127 (staging row)
  const int lc = (tid & 1) * 16;    // 0 or 16 (staging k-offset)

  // Clamp row: out-of-range rows stage duplicate (valid) data; their C rows
  // are never stored, so no load guard is needed anywhere in the main loop.
  const int   grA  = min(rowBase + lr, nrows - 1);
  const float* srcA = X + (size_t)grA * F_IN;
  const float* srcB = W + (size_t)(colBase + lr) * F_IN;

  unsigned int* dstA = (unsigned int*)&lA[lr][lc];
  unsigned int* dstB = (unsigned int*)&lB[lr][lc];

  const int half = lane >> 4;
  const int ml   = lane & 15;

  auto compute = [&]() {
    BF16Frag a[2], b[4];
    #pragma unroll
    for (int ms = 0; ms < 2; ms++) {
      int m = wm * 32 + ms * 16 + ml;
      #pragma unroll
      for (int p = 0; p < 8; p++) {
        int k0 = 2 * p + 16 * half;
        a[ms].w[p] = *(const unsigned int*)&lA[m][k0];
      }
    }
    #pragma unroll
    for (int ns = 0; ns < 4; ns++) {
      int n  = wn * 64 + ns * 16 + ml;
      const uint4* p = (const uint4*)&lB[n][half * 16];
      b[ns].q[0] = p[0];
      b[ns].q[1] = p[1];
    }
    #pragma unroll
    for (int ms = 0; ms < 2; ms++)
      #pragma unroll
      for (int ns = 0; ns < 4; ns++)
        acc[ms][ns] = __builtin_amdgcn_wmma_f32_16x16x32_bf16(
            false, a[ms].v, false, b[ns].v, (short)0, acc[ms][ns], false, false);
  };

  // ---- main loop: kk in [0,480), every 16-elem chunk fully in-bounds
  for (int kk = 0; kk < 480; kk += 32) {
    {
      const float4* s4 = (const float4*)(srcA + kk + lc);
      #pragma unroll
      for (int q = 0; q < 4; q++) {
        float4 x = s4[q];
        dstA[q * 2 + 0] = pack2bf(x.x, x.y);
        dstA[q * 2 + 1] = pack2bf(x.z, x.w);
      }
      __builtin_prefetch(srcA + kk + 32 + lc, 0, 1);
    }
    {
      const float4* s4 = (const float4*)(srcB + kk + lc);
      #pragma unroll
      for (int q = 0; q < 4; q++) {
        float4 x = s4[q];
        dstB[q * 2 + 0] = pack2bf(x.x, x.y);
        dstB[q * 2 + 1] = pack2bf(x.z, x.w);
      }
    }
    __syncthreads();
    compute();
    __syncthreads();
  }

  // ---- peeled K tail (kk = 480): lc==0 chunk full, lc==16 chunk has 4 valid
  if (lc == 0) {
    const float4* sa = (const float4*)(srcA + 480);
    const float4* sb = (const float4*)(srcB + 480);
    #pragma unroll
    for (int q = 0; q < 4; q++) {
      float4 xa = sa[q], xb = sb[q];
      dstA[q * 2 + 0] = pack2bf(xa.x, xa.y);
      dstA[q * 2 + 1] = pack2bf(xa.z, xa.w);
      dstB[q * 2 + 0] = pack2bf(xb.x, xb.y);
      dstB[q * 2 + 1] = pack2bf(xb.z, xb.w);
    }
  } else {
    float4 xa = *(const float4*)(srcA + 496);
    float4 xb = *(const float4*)(srcB + 496);
    dstA[0] = pack2bf(xa.x, xa.y);
    dstA[1] = pack2bf(xa.z, xa.w);
    dstB[0] = pack2bf(xb.x, xb.y);
    dstB[1] = pack2bf(xb.z, xb.w);
    #pragma unroll
    for (int q = 2; q < 8; q++) { dstA[q] = 0u; dstB[q] = 0u; }
  }
  __syncthreads();
  compute();

  // ---- epilogue: C layout lane%16 = N, VGPR v -> M = v + 8*(lane/16)
  #pragma unroll
  for (int ms = 0; ms < 2; ms++)
    #pragma unroll
    for (int ns = 0; ns < 4; ns++)
      #pragma unroll
      for (int v = 0; v < 8; v++) {
        int row = rowBase + wm * 32 + ms * 16 + v + 8 * half;
        int col = colBase + wn * 64 + ns * 16 + ml;
        if (row < nrows) H[(size_t)row * D_HID + col] = acc[ms][ns][v];
      }
}

// ---------------------------------------------------------------- SpMM scatter: dst[row[e],:] += vals[e]*src[col[e],:]
__global__ void spmm_scatter(const int* __restrict__ row, const int* __restrict__ col,
                             const float* __restrict__ vals,
                             const float* __restrict__ src, float* __restrict__ dst) {
  size_t gid = (size_t)blockIdx.x * blockDim.x + threadIdx.x;
  size_t e = gid >> 6;            // 64 threads per edge, 4 floats each
  int    j = (int)(gid & 63);
  if (e >= E_NNZ) return;
  int r = row[e], c = col[e];
  float v = vals[e];
  const float4 x = ((const float4*)(src + (size_t)c * D_HID))[j];
  float* d = dst + (size_t)r * D_HID + j * 4;
  atomicAdd(d + 0, v * x.x);
  atomicAdd(d + 1, v * x.y);
  atomicAdd(d + 2, v * x.z);
  atomicAdd(d + 3, v * x.w);
}

// ---------------------------------------------------------------- selu(h + b1) in-place
__global__ void selu_bias(float* __restrict__ buf, const float* __restrict__ b1) {
  constexpr float kScale = 1.0507009873554805f;
  constexpr float kAlpha = 1.6732632423543772f;
  size_t gid = (size_t)blockIdx.x * blockDim.x + threadIdx.x;
  if (gid >= (size_t)N_NODES * D_HID) return;
  int d = (int)(gid & (D_HID - 1));
  float x = buf[gid] + b1[d];
  buf[gid] = (x > 0.f) ? kScale * x : kScale * kAlpha * (expf(x) - 1.f);
}

// ---------------------------------------------------------------- assignments = softmax(G @ Wt^T + bt) ; 1 wave / 16 nodes
__global__ __launch_bounds__(32) void assign_wmma(
    const float* __restrict__ G, const float* __restrict__ Wt,
    const float* __restrict__ bt, float* __restrict__ S) {
  __shared__ float logit[16][17];
  const int lane = threadIdx.x;
  const int half = lane >> 4;
  const int ml   = lane & 15;
  const int nodeBase = blockIdx.x * 16;   // N divisible by 16

  v8f acc = (v8f){0.f,0.f,0.f,0.f,0.f,0.f,0.f,0.f};
  const float* grow = G  + (size_t)(nodeBase + ml) * D_HID;
  const float* wrow = Wt + (size_t)ml * D_HID;   // cluster ml
  for (int kk = 0; kk < D_HID; kk += 32) {
    BF16Frag a, b;
    #pragma unroll
    for (int p = 0; p < 8; p++) {
      int k0 = 2 * p + 16 * half;
      float2 g2 = *(const float2*)(grow + kk + k0);
      a.w[p] = pack2bf(g2.x, g2.y);
    }
    const float4* w4 = (const float4*)(wrow + kk + half * 16);
    #pragma unroll
    for (int q = 0; q < 4; q++) {
      float4 x = w4[q];
      b.w[q * 2 + 0] = pack2bf(x.x, x.y);
      b.w[q * 2 + 1] = pack2bf(x.z, x.w);
    }
    acc = __builtin_amdgcn_wmma_f32_16x16x32_bf16(
        false, a.v, false, b.v, (short)0, acc, false, false);
  }
  #pragma unroll
  for (int v = 0; v < 8; v++) logit[v + 8 * half][ml] = acc[v] + bt[ml];
  __syncthreads();

  if (lane < 16) {
    int node = nodeBase + lane;
    float mx = -3.0e38f;
    #pragma unroll
    for (int k = 0; k < 16; k++) mx = fmaxf(mx, logit[lane][k]);
    float ex[16]; float sum = 0.f;
    #pragma unroll
    for (int k = 0; k < 16; k++) { ex[k] = expf(logit[lane][k] - mx); sum += ex[k]; }
    float inv = 1.f / sum;
    #pragma unroll
    for (int k = 0; k < 16; k++) S[(size_t)node * K_CL + k] = ex[k] * inv;
  }
}

// ---------------------------------------------------------------- degrees: deg[col[e]] += vals[e]
__global__ void degree_kernel(const int* __restrict__ col, const float* __restrict__ vals,
                              float* __restrict__ deg) {
  size_t e = (size_t)blockIdx.x * blockDim.x + threadIdx.x;
  if (e >= E_NNZ) return;
  atomicAdd(&deg[col[e]], vals[e]);
}

// ---------------------------------------------------------------- trace(S^T A S) = sum_e vals[e] * <S[row_e], S[col_e]>
__global__ void edge_trace(const int* __restrict__ row, const int* __restrict__ col,
                           const float* __restrict__ vals, const float* __restrict__ S,
                           float* __restrict__ acc) {
  size_t e = (size_t)blockIdx.x * blockDim.x + threadIdx.x;
  float partial = 0.f;
  if (e < E_NNZ) {
    const float4* sr = (const float4*)(S + (size_t)row[e] * K_CL);
    const float4* sc = (const float4*)(S + (size_t)col[e] * K_CL);
    float dot = 0.f;
    #pragma unroll
    for (int q = 0; q < 4; q++) {
      float4 a = sr[q], b = sc[q];
      dot += a.x * b.x + a.y * b.y + a.z * b.z + a.w * b.w;
    }
    partial = vals[e] * dot;
  }
  partial = warpSum(partial);
  if ((threadIdx.x & 31) == 0) atomicAdd(acc, partial);
}

// ---------------------------------------------------------------- cluster sizes cs[k]=sum_n S[n,k]; nl[k]=sum_n S[n,k]*deg[n]
__global__ void cluster_stats(const float* __restrict__ S, const float* __restrict__ deg,
                              float* __restrict__ nlbuf, float* __restrict__ csbuf) {
  float lcs[16], lnl[16];
  #pragma unroll
  for (int k = 0; k < 16; k++) { lcs[k] = 0.f; lnl[k] = 0.f; }
  size_t n = (size_t)blockIdx.x * blockDim.x + threadIdx.x;
  size_t stride = (size_t)gridDim.x * blockDim.x;
  for (; n < N_NODES; n += stride) {
    float dg = deg[n];
    const float4* sp = (const float4*)(S + n * K_CL);
    #pragma unroll
    for (int q = 0; q < 4; q++) {
      float4 s = sp[q];
      lcs[q*4+0] += s.x; lcs[q*4+1] += s.y; lcs[q*4+2] += s.z; lcs[q*4+3] += s.w;
      lnl[q*4+0] += s.x*dg; lnl[q*4+1] += s.y*dg; lnl[q*4+2] += s.z*dg; lnl[q*4+3] += s.w*dg;
    }
  }
  #pragma unroll
  for (int k = 0; k < 16; k++) {
    float r = warpSum(lcs[k]);
    if ((threadIdx.x & 31) == 0) atomicAdd(&csbuf[k], r);
    r = warpSum(lnl[k]);
    if ((threadIdx.x & 31) == 0) atomicAdd(&nlbuf[k], r);
  }
}

// ---------------------------------------------------------------- per-column max of G  (thread = column, coalesced)
__global__ __launch_bounds__(256) void colmax_kernel(const float* __restrict__ G,
                                                     float* __restrict__ colmax) {
  const int d = threadIdx.x;
  int r0 = blockIdx.x * 256;
  int r1 = min(r0 + 256, N_NODES);
  float m = -3.0e38f;
  for (int r = r0; r < r1; r++) m = fmaxf(m, G[(size_t)r * D_HID + d]);
  atomicMaxF(&colmax[d], m);
}

// ---------------------------------------------------------------- per-column sumexp / sum(aug) / sum(aug*G)
__global__ __launch_bounds__(256) void colstats_kernel(
    const float* __restrict__ G, const float* __restrict__ AUG,
    const float* __restrict__ colmax, float* __restrict__ sumexp,
    float* __restrict__ sumaug, float* __restrict__ sumprod) {
  const int d = threadIdx.x;
  int r0 = blockIdx.x * 256;
  int r1 = min(r0 + 256, N_NODES);
  float m = colmax[d];
  float se = 0.f, sa = 0.f, sp = 0.f;
  for (int r = r0; r < r1; r++) {
    float g = G[(size_t)r * D_HID + d];
    float a = AUG[(size_t)r * D_HID + d];
    se += expf(g - m);
    sa += a;
    sp += a * g;
  }
  atomicAdd(&sumexp[d], se);
  atomicAdd(&sumaug[d], sa);
  atomicAdd(&sumprod[d], sp);
}

// ---------------------------------------------------------------- combine all losses
__global__ __launch_bounds__(256) void finalize_kernel(
    const float* __restrict__ colmax, const float* __restrict__ sumexp,
    const float* __restrict__ sumaug, const float* __restrict__ sumprod,
    const float* __restrict__ nlbuf, const float* __restrict__ csbuf,
    const float* __restrict__ traceAcc, float* __restrict__ out) {
  __shared__ float red[256];
  const int tid = threadIdx.x;
  float lse = colmax[tid] + logf(sumexp[tid]);
  red[tid] = sumprod[tid] - lse * sumaug[tid];
  __syncthreads();
  for (int s = 128; s > 0; s >>= 1) {
    if (tid < s) red[tid] += red[tid + s];
    __syncthreads();
  }
  if (tid == 0) {
    const float Ef = (float)E_NNZ;
    float conSum = red[0];
    float nlsq = 0.f, cssq = 0.f;
    for (int k = 0; k < K_CL; k++) {
      nlsq += nlbuf[k] * nlbuf[k];
      cssq += csbuf[k] * csbuf[k];
    }
    float traceAS   = traceAcc[0];
    float spectral  = -(traceAS - nlsq / (2.f * Ef)) / (2.f * Ef);
    float cluster   = (sqrtf(cssq) / (float)N_NODES * 4.0f - 1.0f) * 1.0f;  // sqrt(K)=4, CLUSTER_REG=1
    float con       = -conSum / (float)D_HID;
    out[0] = spectral + cluster + 0.5f * con;                               // CON_REG=0.5
  }
}

// ================================================================ host
extern "C" void kernel_launch(void* const* d_in, const int* in_sizes, int n_in,
                              void* d_out, int out_size, void* d_ws, size_t ws_size,
                              hipStream_t stream) {
  const float* features     = (const float*)d_in[0];
  const float* aug_features = (const float*)d_in[1];
  const int*   graph_row    = (const int*)d_in[2];
  const int*   graph_col    = (const int*)d_in[3];
  const float* graph_vals   = (const float*)d_in[4];
  const float* gn_vals      = (const float*)d_in[5];
  const float* W1           = (const float*)d_in[8];
  const float* b1           = (const float*)d_in[9];
  const float* Wt           = (const float*)d_in[10];
  const float* bt           = (const float*)d_in[11];
  float* out = (float*)d_out;

  const size_t ND = (size_t)N_NODES * D_HID;
  float* ws     = (float*)d_ws;
  float* H1     = ws;
  float* G      = ws + ND;
  float* AUG    = ws + 2 * ND;
  float* S      = ws + 3 * ND;
  float* deg    = S + (size_t)N_NODES * K_CL;
  float* colmax = deg + N_NODES;
  float* sumexp = colmax + D_HID;
  float* sumaug = sumexp + D_HID;
  float* sumprod= sumaug + D_HID;
  float* nlbuf  = sumprod + D_HID;
  float* csbuf  = nlbuf + K_CL;
  float* acc    = csbuf + K_CL;

  // init accumulators
  fill_kernel<<<2048, 256, 0, stream>>>(G, ND, 0.f);
  fill_kernel<<<2048, 256, 0, stream>>>(AUG, ND, 0.f);
  fill_kernel<<<256, 256, 0, stream>>>(deg, (size_t)N_NODES, 0.f);
  fill_kernel<<<1, 256, 0, stream>>>(colmax, (size_t)D_HID, -3.0e38f);
  fill_kernel<<<4, 256, 0, stream>>>(sumexp, (size_t)(3 * D_HID + 2 * K_CL + 1), 0.f);

  dim3 gemmGrid((N_NODES + 127) / 128, D_HID / 128);

  // ---- gcn(features)
  gemm_bf16_wmma<<<gemmGrid, 256, 0, stream>>>(features, W1, H1, N_NODES);
  spmm_scatter<<<(int)(((size_t)E_NNZ * 64) / 256), 256, 0, stream>>>(
      graph_row, graph_col, gn_vals, H1, G);
  selu_bias<<<(int)(ND / 256), 256, 0, stream>>>(G, b1);

  // ---- gcn(aug_features) (H1 reused)
  gemm_bf16_wmma<<<gemmGrid, 256, 0, stream>>>(aug_features, W1, H1, N_NODES);
  spmm_scatter<<<(int)(((size_t)E_NNZ * 64) / 256), 256, 0, stream>>>(
      graph_row, graph_col, gn_vals, H1, AUG);
  selu_bias<<<(int)(ND / 256), 256, 0, stream>>>(AUG, b1);

  // ---- assignments + graph statistics
  assign_wmma<<<N_NODES / 16, 32, 0, stream>>>(G, Wt, bt, S);
  degree_kernel<<<(E_NNZ + 255) / 256, 256, 0, stream>>>(graph_col, graph_vals, deg);
  edge_trace<<<(E_NNZ + 255) / 256, 256, 0, stream>>>(graph_row, graph_col, graph_vals, S, acc);
  cluster_stats<<<128, 256, 0, stream>>>(S, deg, nlbuf, csbuf);

  // ---- contrastive column stats
  colmax_kernel<<<(N_NODES + 255) / 256, 256, 0, stream>>>(G, colmax);
  colstats_kernel<<<(N_NODES + 255) / 256, 256, 0, stream>>>(G, AUG, colmax, sumexp, sumaug, sumprod);

  // ---- combine
  finalize_kernel<<<1, 256, 0, stream>>>(colmax, sumexp, sumaug, sumprod, nlbuf, csbuf, acc, out);
}
